// DeformConvNet_74036646248776
// MI455X (gfx1250) — matrix operand order repaired and checked by reading, not compiled, not run on hardware
//
#include <hip/hip_runtime.h>

#define B_    8
#define C_    128
#define H_    128
#define W_    128
#define HW_   (H_ * W_)        // 16384
#define KSZ   1152             // C_ * 9
#define NTOT  (B_ * HW_)       // 131072
#define OC1   256
#define OC2   128
#define BN_EPS 1e-5f

typedef __attribute__((ext_vector_type(16))) _Float16 v16h;
typedef __attribute__((ext_vector_type(8)))  float    v8f;

// ---------------- fp32 -> f16 weight conversion ----------------
__global__ void cvt_f32_to_f16(const float* __restrict__ src,
                               _Float16* __restrict__ dst, int n) {
  int i = blockIdx.x * blockDim.x + threadIdx.x;
  if (i < n) dst[i] = (_Float16)src[i];
}

// ---------------- implicit-GEMM 3x3 conv via WMMA ----------------
// MODE 0: plain f32 store (conv1 -> offsets buffer)
// MODE 1: + bias + ReLU (conv2 -> d_out)
// Workgroup: 256 threads (8 waves). Tile: 128 OC rows x 128 spatial cols (one image row).
// A-fragments stream from L2 (weights hot; each byte used once per tile -> no LDS value);
// LDS holds only a 16KB double-buffered im2col B tile => many WGs resident per WGP.
// k-loop unrolled by 2 with A0/A1 register ping-pong: no A copy, no WMMA->VALU hazard NOPs.
template <int MODE, typename SrcT>
__global__ void __launch_bounds__(256)
conv3x3_wmma(const SrcT* __restrict__ src,      // (B, C, H, W)
             const _Float16* __restrict__ w16,  // (oc_total, 1152)
             const float* __restrict__ bias,    // (oc_total) for MODE 1
             float* __restrict__ out,           // (B, oc_total, H, W)
             int oc_total) {
  __shared__ _Float16 Bbuf[2 * 4096];            // 2 x (32K x 128N) halfs = 16KB

  const int tid  = threadIdx.x;
  const int wave = tid >> 5;
  const int lane = tid & 31;
  const int oc0  = blockIdx.y * 128;

  // ---- spatial tile: one full image row (128 cols) ----
  const int gn0  = blockIdx.x * 128;
  const int bimg = gn0 >> 14;                    // / HW_
  const int i0   = (gn0 & (HW_ - 1)) >> 7;       // image row

  v8f acc[8] = {{}, {}, {}, {}, {}, {}, {}, {}};

  // ---- B-fill role: one k-row (of 32), 16 consecutive n cols (one 16-block) ----
  const int kf    = tid >> 3;                    // 0..31
  const int nf0   = (tid & 7) * 16;              // 0,16,...,112
  const int laneo = (kf >= 16) ? 16 : 0;
  const int halfo = kf & 15;
  // fragment-layout destination base (buffer 0); element nn => +16 halfs
  _Float16* dst0 = Bbuf + (((nf0 >> 4) * 32 + laneo) << 4) + halfo;

  // incremental decomposition of kk = kb*32 + kf into (ic, r, s)
  int ic  = kf / 9;
  int tap = kf - ic * 9;
  int r   = tap / 3;
  int s   = tap - r * 3;

  float vals[16];
  auto fetch = [&](int icc, int rr, int ss) {
    int y   = i0 + rr - 1;
    int xc0 = nf0 + ss - 1;
    const SrcT* row = src + ((size_t)(bimg * C_ + icc) * H_ + y) * W_;
    if ((unsigned)y < (unsigned)H_) {
      if (xc0 >= 0 && xc0 <= W_ - 16) {
        // interior fast path: unconditional contiguous loads
#pragma unroll
        for (int nn = 0; nn < 16; ++nn) vals[nn] = (float)row[xc0 + nn];
      } else {
        // branchless edge: clamped address + select
#pragma unroll
        for (int nn = 0; nn < 16; ++nn) {
          int xc  = xc0 + nn;
          int xcc = min(max(xc, 0), W_ - 1);
          float v = (float)row[xcc];
          vals[nn] = (xc == xcc) ? v : 0.f;
        }
      }
    } else {
#pragma unroll
      for (int nn = 0; nn < 16; ++nn) vals[nn] = 0.f;
    }
  };

  // ---- A fragments stream straight from global (L2-hot weights) ----
  const int kb8  = (lane < 16) ? 0 : 16;
  const int mrow = (wave << 4) + (lane & 15);
  const _Float16* arow = w16 + (size_t)(oc0 + mrow) * KSZ + kb8;

  union AB { v16h v; uint4 q[2]; };
  AB A0, A1;
  auto loadA = [&](int kb, AB& A) {
    const _Float16* p = arow + kb * 32;
    A.q[0] = *(const uint4*)(p);
    A.q[1] = *(const uint4*)(p + 8);
  };
  loadA(0, A0);

  // prologue: fill buffer 0 with k-block 0
  fetch(ic, r, s);
#pragma unroll
  for (int nn = 0; nn < 16; ++nn) dst0[nn * 16] = (_Float16)vals[nn];
  __syncthreads();

  // one pipeline step: WMMA on k-block kb using Ac; prefetch kb+1 into An + LDS
  auto step = [&](int kb, AB& Ac, AB& An, bool more) {
    int ic_n = ic, r_n = r, s_n = s;
    if (more) {
      s_n = s + 2; r_n = r + 1; ic_n = ic + 3;       // kk += 32
      if (s_n >= 3) { s_n -= 3; r_n += 1; }
      if (r_n >= 3) { r_n -= 3; ic_n += 1; }
      fetch(ic_n, r_n, s_n);
      loadA(kb + 1, An);
    }
    const _Float16* Bf = Bbuf + (kb & 1) * 4096;
#pragma unroll
    for (int nb = 0; nb < 8; ++nb) {
      AB Bm;
      const _Float16* bb = Bf + ((nb * 32 + lane) << 4);
      Bm.q[0] = *(const uint4*)(bb);
      Bm.q[1] = *(const uint4*)(bb + 8);
      acc[nb] = __builtin_amdgcn_wmma_f32_16x16x32_f16(
          false, Ac.v, false, Bm.v, (short)0, acc[nb], false, false);
    }
    if (more) {
      _Float16* d = dst0 + ((kb + 1) & 1) * 4096;
#pragma unroll
      for (int nn = 0; nn < 16; ++nn) d[nn * 16] = (_Float16)vals[nn];
      ic = ic_n; r = r_n; s = s_n;
    }
    __syncthreads();
  };

  for (int kb2 = 0; kb2 < (KSZ / 32) / 2; ++kb2) {
    step(2 * kb2,     A0, A1, true);                      // kb+1 <= 35 always valid
    step(2 * kb2 + 1, A1, A0, 2 * kb2 + 2 < KSZ / 32);
  }

  // ---- epilogue: D layout = lanes 0-15 N=lane M=v, lanes 16-31 N=lane-16 M=v+8 ----
  const int mbase = oc0 + (wave << 4) + ((lane < 16) ? 0 : 8);
  const int ncol  = lane & 15;
  const int rembase = i0 * 128 + ncol;

  if (MODE == 0) {
#pragma unroll
    for (int nb = 0; nb < 8; ++nb) {
      float* op = out + ((size_t)bimg * oc_total + mbase) * HW_ + rembase + nb * 16;
#pragma unroll
      for (int vv = 0; vv < 8; ++vv) op[(size_t)vv * HW_] = acc[nb][vv];
    }
  } else {
    float bv[8];
#pragma unroll
    for (int vv = 0; vv < 8; ++vv) bv[vv] = bias[mbase + vv];
#pragma unroll
    for (int nb = 0; nb < 8; ++nb) {
      float* op = out + ((size_t)bimg * oc_total + mbase) * HW_ + rembase + nb * 16;
#pragma unroll
      for (int vv = 0; vv < 8; ++vv) {
        float val = fmaxf(acc[nb][vv] + bv[vv], 0.f);
        op[(size_t)vv * HW_] = val;
      }
    }
  }
}

// ---------------- deformable bilinear sampling ----------------
// Faithful to the reference's raw reshape (b,2c,h,w)->(b*c,h*w,2):
// for image-plane p=b*C+c at grid point q: oy = raw[p*2HW + 2q], ox = raw[p*2HW + 2q+1]
__global__ void deform_sample(const float* __restrict__ x,
                              const float* __restrict__ off,
                              _Float16* __restrict__ xoff) {
  int idx = blockIdx.x * blockDim.x + threadIdx.x;
  if (idx >= B_ * C_ * HW_) return;
  int p = idx >> 14;
  int q = idx & (HW_ - 1);
  float2 o2 = *(const float2*)(off + (size_t)p * (2 * HW_) + 2 * q);
  int i = q >> 7, j = q & 127;
  float yc = fminf(fmaxf(o2.x + (float)i, 0.f), (float)(H_ - 1));
  float xc = fminf(fmaxf(o2.y + (float)j, 0.f), (float)(W_ - 1));
  float y0f = floorf(yc), y1f = ceilf(yc);
  float x0f = floorf(xc), x1f = ceilf(xc);
  int y0 = (int)y0f, y1 = (int)y1f, x0 = (int)x0f, x1 = (int)x1f;
  const float* img = x + (size_t)p * HW_;
  float v_lt = img[y0 * W_ + x0];
  float v_rb = img[y1 * W_ + x1];
  float v_lb = img[y0 * W_ + x1];
  float v_rt = img[y1 * W_ + x0];
  float dy = yc - y0f, dx = xc - x0f;
  float v_t = v_lt + (v_rt - v_lt) * dy;
  float v_b = v_lb + (v_rb - v_lb) * dy;
  xoff[idx] = (_Float16)(v_t + (v_b - v_t) * dx);
}

// ---------------- deterministic BN statistics (per-channel tree reduce) ----------------
__global__ void bn_stats(const float* __restrict__ y, float* __restrict__ stats) {
  __shared__ float ssum[256];
  __shared__ float ssq[256];
  int ch  = blockIdx.x;
  int tid = threadIdx.x;
  float s = 0.f, q = 0.f;
  for (int b = 0; b < B_; ++b) {
    const float* p = y + ((size_t)b * C_ + ch) * HW_;
    for (int i = tid * 4; i < HW_; i += 1024) {
      float4 v = *(const float4*)(p + i);
      s += v.x + v.y + v.z + v.w;
      q += v.x * v.x + v.y * v.y + v.z * v.z + v.w * v.w;
    }
  }
  ssum[tid] = s;
  ssq[tid]  = q;
  __syncthreads();
  for (int st = 128; st > 0; st >>= 1) {
    if (tid < st) { ssum[tid] += ssum[tid + st]; ssq[tid] += ssq[tid + st]; }
    __syncthreads();
  }
  if (tid == 0) {
    stats[ch]       = ssum[0];
    stats[128 + ch] = ssq[0];
  }
}

__global__ void bn_finalize(float* __restrict__ y,
                            const float* __restrict__ stats,
                            const float* __restrict__ gamma,
                            const float* __restrict__ beta) {
  int idx4 = blockIdx.x * blockDim.x + threadIdx.x;
  if (idx4 >= (B_ * C_ * HW_) / 4) return;
  size_t base = (size_t)idx4 * 4;
  int ch = (int)((base >> 14) & (C_ - 1));
  const float invN = 1.0f / (float)(B_ * HW_);
  float mean = stats[ch] * invN;
  float var  = stats[128 + ch] * invN - mean * mean;
  float scl  = gamma[ch] * rsqrtf(var + BN_EPS);
  float sft  = beta[ch] - mean * scl;
  float4 v = *(float4*)(y + base);
  v.x = v.x * scl + sft;
  v.y = v.y * scl + sft;
  v.z = v.z * scl + sft;
  v.w = v.w * scl + sft;
  *(float4*)(y + base) = v;
}

// ---------------- launcher ----------------
extern "C" void kernel_launch(void* const* d_in, const int* in_sizes, int n_in,
                              void* d_out, int out_size, void* d_ws, size_t ws_size,
                              hipStream_t stream) {
  (void)in_sizes; (void)n_in; (void)out_size; (void)ws_size;
  const float* x      = (const float*)d_in[0];
  const float* w_off  = (const float*)d_in[1];
  const float* w_conv = (const float*)d_in[2];
  const float* b_conv = (const float*)d_in[3];
  const float* gamma  = (const float*)d_in[4];
  const float* beta   = (const float*)d_in[5];
  float* out = (float*)d_out;

  // workspace layout
  char* ws = (char*)d_ws;
  float*    off_f32 = (float*)ws;                                   // 134,217,728 B
  _Float16* xoff16  = (_Float16*)(ws + 134217728);                  //  33,554,432 B
  _Float16* woff16  = (_Float16*)(ws + 134217728 + 33554432);       //     589,824 B
  _Float16* wconv16 = (_Float16*)(ws + 134217728 + 33554432 + 589824);       // 294,912 B
  float*    stats   = (float*)(ws + 134217728 + 33554432 + 589824 + 294912); //   1,024 B

  // 1) weights -> f16
  cvt_f32_to_f16<<<(OC1 * KSZ + 255) / 256, 256, 0, stream>>>(w_off, woff16, OC1 * KSZ);
  cvt_f32_to_f16<<<(OC2 * KSZ + 255) / 256, 256, 0, stream>>>(w_conv, wconv16, OC2 * KSZ);

  // 2) offsets = conv3x3(x, w_off)   (M=256 -> grid.y=2)
  {
    dim3 g(NTOT / 128, 2);
    conv3x3_wmma<0, float><<<g, 256, 0, stream>>>(x, woff16, nullptr, off_f32, OC1);
  }

  // 3) deformable bilinear sampling -> x_off (f16)
  deform_sample<<<(B_ * C_ * HW_ + 255) / 256, 256, 0, stream>>>(x, off_f32, xoff16);

  // 4) y = relu(conv3x3(x_off, w_conv) + b) -> d_out
  {
    dim3 g(NTOT / 128, 1);
    conv3x3_wmma<1, _Float16><<<g, 256, 0, stream>>>(xoff16, wconv16, b_conv, out, OC2);
  }

  // 5) batch-norm (training mode, biased variance), deterministic reduction
  bn_stats<<<C_, 256, 0, stream>>>(out, stats);
  bn_finalize<<<((B_ * C_ * HW_ / 4) + 255) / 256, 256, 0, stream>>>(out, stats, gamma, beta);
}
